// VQ_VAE_9844065042630
// MI455X (gfx1250) — matrix operand order repaired and checked by reading, not compiled, not run on hardware
//
#include <hip/hip_runtime.h>

// ---------------------------------------------------------------------------
// VQ-VAE forward on gfx1250 (CDNA5). All conv / deconv / VQ-distance math runs
// through v_wmma_f32_16x16x32_f16 (f16 operands, f32 accumulate).
// One wave (32 lanes) owns a 16(M pixels) x 16*NT(N couts) output tile.
// Weights / codebook are pre-packed to f16 K-contiguous layouts so every
// fragment is fetched with global_load_b128 pairs.
// ---------------------------------------------------------------------------

typedef _Float16 v16h __attribute__((ext_vector_type(16)));
typedef _Float16 v8h  __attribute__((ext_vector_type(8)));
typedef float    v8f  __attribute__((ext_vector_type(8)));

static __device__ __forceinline__ v16h cat8(v8h lo, v8h hi) {
  return __builtin_shufflevector(lo, hi, 0, 1, 2, 3, 4, 5, 6, 7,
                                 8, 9, 10, 11, 12, 13, 14, 15);
}

// -------------------------- helper kernels ---------------------------------

__global__ void k_f32_to_f16(const float* __restrict__ in,
                             _Float16* __restrict__ out, int n) {
  int i = blockIdx.x * blockDim.x + threadIdx.x;
  if (i < n) out[i] = (_Float16)in[i];
}

// pack w[tap][ci][co] (f32) -> wp[tap][co][ci] (f16); also used to transpose
// the codebook cb[D][K] -> cbT[K][D] with taps=1.
__global__ void k_pack_w(const float* __restrict__ w, _Float16* __restrict__ wp,
                         int taps, int Cin, int Cout) {
  int t = blockIdx.x * blockDim.x + threadIdx.x;
  int total = taps * Cin * Cout;
  if (t >= total) return;
  int tap = t / (Cin * Cout);
  int r   = t - tap * (Cin * Cout);
  int co  = r / Cin;
  int ci  = r - co * Cin;
  wp[t] = (_Float16)w[(size_t)(tap * Cin + ci) * Cout + co];
}

// sum of squares of each contiguous row (one wave per row)
__global__ void k_row_sumsq(const float* __restrict__ in,
                            float* __restrict__ out, int len) {
  int row = blockIdx.x;
  const float* p = in + (long)row * len;
  float s = 0.f;
  for (int i = threadIdx.x; i < len; i += 32) { float v = p[i]; s += v * v; }
  #pragma unroll
  for (int m = 16; m >= 1; m >>= 1) s += __shfl_xor(s, m, 32);
  if (threadIdx.x == 0) out[row] = s;
}

// sum of squares of each column of cb[D][K]
__global__ void k_col_sumsq(const float* __restrict__ cb,
                            float* __restrict__ out, int D, int K) {
  int n = blockIdx.x * blockDim.x + threadIdx.x;
  if (n >= K) return;
  float s = 0.f;
  for (int d = 0; d < D; ++d) { float v = cb[(long)d * K + n]; s += v * v; }
  out[n] = s;
}

// ------------------- unified implicit-GEMM conv kernel ----------------------
// Normal conv (lhsdil=1, stride s, pad p) and conv_transpose (stride=1 conv
// over lhs-dilated input; jax 'SAME' transpose padding; kernel NOT flipped).
// Grid: (Mtiles, Ntiles/NT), block = one wave (32). A fragment is reused
// across NT output-channel tiles.
template <int CIN, int NT>
__global__ void k_conv_wmma(const _Float16* __restrict__ in,   // NHWC f16
                            const _Float16* __restrict__ wp,   // [tap][Cout][CIN] f16
                            const float* __restrict__ bias,    // [Cout]
                            _Float16* __restrict__ outH,       // nullable
                            float* __restrict__ outF,          // nullable
                            int B, int H, int W,
                            int Hout, int Wout, int Cout,
                            int KH, int KW, int stride, int pad,
                            int lhsdil, int relu) {
  const int lane  = threadIdx.x;
  const int ml    = lane & 15;   // row (A) / col (B) within tile
  const int hl    = lane >> 4;   // K-half selector
  const int mBase = blockIdx.x * 16;
  const int nBase = blockIdx.y * (16 * NT);
  const v16h vz16 = {};

  // decode the output pixel this lane feeds into the A fragment
  const int hw  = Hout * Wout;
  const int p   = mBase + ml;
  const int b   = p / hw;
  const int rem = p - b * hw;
  const int oy  = rem / Wout;
  const int ox  = rem - oy * Wout;
  const long inBase = (long)b * H * W * CIN;

  const int co = nBase + ml;     // B-fragment base column for this lane

  v8f acc[NT] = {};

  for (int ky = 0; ky < KH; ++ky) {
    int qy = oy * stride + ky - pad;
    int iy = (lhsdil == 2) ? (qy >> 1) : qy;
    bool vy = (qy >= 0) && (iy < H) && ((lhsdil == 2) ? ((qy & 1) == 0) : true);
    for (int kx = 0; kx < KW; ++kx) {
      int qx = ox * stride + kx - pad;
      int ix = (lhsdil == 2) ? (qx >> 1) : qx;
      bool vx = (qx >= 0) && (ix < W) && ((lhsdil == 2) ? ((qx & 1) == 0) : true);
      const bool valid = vy && vx;
      if (!__any(valid)) continue;   // whole tap dead for this tile (deconv parity)
      const _Float16* aptr = in + inBase + ((long)iy * W + ix) * CIN;
      const _Float16* wtap = wp + (long)(ky * KW + kx) * Cout * CIN;
      __builtin_prefetch(wtap + (long)co * CIN, 0, 0);  // global_prefetch_b8
      #pragma unroll
      for (int c0 = 0; c0 < CIN; c0 += 32) {
        v16h a;
        if constexpr (CIN % 32 == 0) {
          // A: lanes 0-15 hold K 0..15, lanes 16-31 hold K 16..31
          if (valid) {
            a = cat8(*(const v8h*)(aptr + c0 + hl * 16),
                     *(const v8h*)(aptr + c0 + hl * 16 + 8));
          } else {
            a = vz16;
          }
        } else {
          #pragma unroll
          for (int e = 0; e < 16; ++e) {
            int ka = e + hl * 16;
            int ca = c0 + ka;
            a[e] = (valid && ca < CIN) ? aptr[ca] : (_Float16)0.f;
          }
        }
        #pragma unroll
        for (int t = 0; t < NT; ++t) {
          const int cot = co + 16 * t;
          const _Float16* wrow = wtap + (long)cot * CIN;
          v16h bf;
          if constexpr (CIN % 32 == 0) {
            // B: lanes 0-15 hold K 0..15, lanes 16-31 hold K 16..31
            if (cot < Cout) {
              bf = cat8(*(const v8h*)(wrow + c0 + hl * 16),
                        *(const v8h*)(wrow + c0 + hl * 16 + 8));
            } else {
              bf = vz16;
            }
          } else {
            #pragma unroll
            for (int e = 0; e < 16; ++e) {
              int cbk = c0 + e + hl * 16;
              bf[e] = (cbk < CIN && cot < Cout) ? wrow[cbk] : (_Float16)0.f;
            }
          }
          acc[t] = __builtin_amdgcn_wmma_f32_16x16x32_f16(
              false, a, false, bf, (short)0, acc[t], false, false);
        }
      }
    }
  }

  // epilogue: lane (n = ml), VGPR j -> row m = j + hl*8
  #pragma unroll
  for (int t = 0; t < NT; ++t) {
    const int cot = co + 16 * t;
    if (cot >= Cout) continue;
    float bv = bias[cot];
    #pragma unroll
    for (int j = 0; j < 8; ++j) {
      int m  = j + hl * 8;
      long o = (long)(mBase + m) * Cout + cot;
      float v = acc[t][j] + bv;
      if (relu) v = v > 0.f ? v : 0.f;
      if (outH) outH[o] = (_Float16)v;
      if (outF) outF[o] = v;
    }
  }
}

// ------------------------- vector quantizer ---------------------------------
// dist[m][n] = ||z_m||^2 + ||e_n||^2 - 2 (z_m . e_n); argmin over n, then
// gather codebook rows (pre-transposed cbT[K][D] f16) as the decoder input.
// All 16 A (z) fragments are held in VGPRs (128 regs; launch_bounds(32,1)
// gives the wave the whole register file so nothing spills) and reused across
// all 64 codebook tiles, 4 tiles per pass.
__global__ void __launch_bounds__(32, 1)
k_vq_wmma(const _Float16* __restrict__ z16,  // [4096][512]
          const _Float16* __restrict__ cbT,  // [1024][512]
          const float* __restrict__ fn2,     // [4096]
          const float* __restrict__ en2,     // [1024]
          _Float16* __restrict__ quant,      // [4096][512]
          int* __restrict__ idxOut) {        // [4096]
  const int D = 512, K = 1024;
  const int lane  = threadIdx.x;
  const int ml    = lane & 15;
  const int hl    = lane >> 4;
  const int mBase = blockIdx.x * 16;
  __shared__ int sIdx[16];

  const _Float16* zrow = z16 + (long)(mBase + ml) * D;

  // resident A fragments: 16 chunks x 8 VGPRs
  v16h aFrag[16];
  #pragma unroll
  for (int c = 0; c < 16; ++c) {
    aFrag[c] = cat8(*(const v8h*)(zrow + c * 32 + hl * 16),
                    *(const v8h*)(zrow + c * 32 + hl * 16 + 8));
  }

  float rn[8];
  #pragma unroll
  for (int j = 0; j < 8; ++j) rn[j] = fn2[mBase + j + hl * 8];

  float minv[8]; int mini[8];
  #pragma unroll
  for (int j = 0; j < 8; ++j) { minv[j] = 3.4e38f; mini[j] = 0; }

  for (int ntg = 0; ntg < K / 64; ++ntg) {   // 4 codebook tiles per pass
    const int co0 = ntg * 64 + ml;
    v8f acc[4] = {};
    #pragma unroll
    for (int c = 0; c < 16; ++c) {
      #pragma unroll
      for (int t = 0; t < 4; ++t) {
        const _Float16* crow = cbT + (long)(co0 + 16 * t) * D + c * 32;
        v16h bf = cat8(*(const v8h*)(crow + hl * 16),
                       *(const v8h*)(crow + hl * 16 + 8));
        acc[t] = __builtin_amdgcn_wmma_f32_16x16x32_f16(
            false, aFrag[c], false, bf, (short)0, acc[t], false, false);
      }
    }
    #pragma unroll
    for (int t = 0; t < 4; ++t) {
      const int cot = co0 + 16 * t;
      float en = en2[cot];
      #pragma unroll
      for (int j = 0; j < 8; ++j) {
        float d = rn[j] + en - 2.f * acc[t][j];
        if (d < minv[j]) { minv[j] = d; mini[j] = cot; }
      }
    }
  }

  // argmin reduce across the 16 lanes of each half (xor masks stay in-group)
  #pragma unroll
  for (int off = 1; off < 16; off <<= 1) {
    #pragma unroll
    for (int j = 0; j < 8; ++j) {
      float ov = __shfl_xor(minv[j], off, 32);
      int   oi = __shfl_xor(mini[j], off, 32);
      if (ov < minv[j] || (ov == minv[j] && oi < mini[j])) {
        minv[j] = ov; mini[j] = oi;
      }
    }
  }
  if (ml == 0) {
    #pragma unroll
    for (int j = 0; j < 8; ++j) {
      int r = j + hl * 8;
      sIdx[r] = mini[j];
      idxOut[mBase + r] = mini[j];
    }
  }
  __syncthreads();

  // gather (vectorized): quantized[r][:] = cbT[idx[r]][:]
  const int DV = D / 8;                       // v8h chunks per row
  for (int t = lane; t < 16 * DV; t += 32) {
    int r  = t / DV;
    int d8 = t - r * DV;
    int id = sIdx[r];
    ((v8h*)(quant + (long)(mBase + r) * D))[d8] =
        ((const v8h*)(cbT + (long)id * D))[d8];
  }
}

// ------------------------------ launcher ------------------------------------

extern "C" void kernel_launch(void* const* d_in, const int* in_sizes, int n_in,
                              void* d_out, int out_size, void* d_ws, size_t ws_size,
                              hipStream_t stream) {
  (void)in_sizes; (void)n_in; (void)out_size; (void)ws_size;

  const float* x   = (const float*)d_in[0];
  const float* ew1 = (const float*)d_in[1];  const float* eb1 = (const float*)d_in[2];
  const float* ew2 = (const float*)d_in[3];  const float* eb2 = (const float*)d_in[4];
  const float* ew3 = (const float*)d_in[5];  const float* eb3 = (const float*)d_in[6];
  const float* ew4 = (const float*)d_in[7];  const float* eb4 = (const float*)d_in[8];
  const float* ew5 = (const float*)d_in[9];  const float* eb5 = (const float*)d_in[10];
  const float* dw1 = (const float*)d_in[11]; const float* db1 = (const float*)d_in[12];
  const float* dw2 = (const float*)d_in[13]; const float* db2 = (const float*)d_in[14];
  const float* dw3 = (const float*)d_in[15]; const float* db3 = (const float*)d_in[16];
  const float* dw4 = (const float*)d_in[17]; const float* db4 = (const float*)d_in[18];
  const float* dw5 = (const float*)d_in[19]; const float* db5 = (const float*)d_in[20];
  const float* cb  = (const float*)d_in[21];

  const int Bn = 256;

  // workspace layout
  char* ws = (char*)d_ws;
  _Float16* P0  = (_Float16*)ws;                            // <= 67.2 MB (d4 act)
  _Float16* P1  = (_Float16*)(ws + ((size_t)72 << 20));     // <= 33.6 MB
  float*    ZB  = (float*)(ws + ((size_t)144 << 20));       // 8.4 MB (z f32)
  float*    FN2 = (float*)(ws + ((size_t)154 << 20));       // 4096 floats
  float*    EN2 = FN2 + 4096;                               // 1024 floats
  _Float16* WP  = (_Float16*)(ws + ((size_t)155 << 20));    // packed weights <=2.4MB
  _Float16* CBT = (_Float16*)(ws + ((size_t)160 << 20));    // cb^T f16, 1 MB
  _Float16* Z16 = (_Float16*)(ws + ((size_t)163 << 20));    // z f16, 4.2 MB

  float* recon = (float*)d_out;                             // 256*64*64*3 f32
  int*   idxO  = (int*)(recon + (size_t)Bn * 64 * 64 * 3);  // 4096 int32

  auto packw = [&](const float* w, int taps, int ci, int co) {
    int n = taps * ci * co;
    k_pack_w<<<(n + 255) / 256, 256, 0, stream>>>(w, WP, taps, ci, co);
  };

  // 0) x f32 -> f16
  {
    int n = Bn * 64 * 64 * 3;
    k_f32_to_f16<<<(n + 255) / 256, 256, 0, stream>>>(x, P0, n);
  }

  // --- encoder (stride-2 'SAME' convs: pad_before = 0) ---
  packw(ew1, 9, 3, 32);
  k_conv_wmma<3, 2><<<dim3(Bn*32*32/16, 1), 32, 0, stream>>>(P0, WP, eb1, P1, nullptr,
      Bn, 64, 64, 32, 32,  32, 3, 3, 2, 0, 1, 1);
  packw(ew2, 9, 32, 64);
  k_conv_wmma<32, 2><<<dim3(Bn*16*16/16, 2), 32, 0, stream>>>(P1, WP, eb2, P0, nullptr,
      Bn, 32, 32, 16, 16,  64, 3, 3, 2, 0, 1, 1);
  packw(ew3, 9, 64, 128);
  k_conv_wmma<64, 2><<<dim3(Bn*8*8/16, 4), 32, 0, stream>>>(P0, WP, eb3, P1, nullptr,
      Bn, 16, 16,  8,  8, 128, 3, 3, 2, 0, 1, 1);
  packw(ew4, 9, 128, 256);
  k_conv_wmma<128, 2><<<dim3(Bn*4*4/16, 8), 32, 0, stream>>>(P1, WP, eb4, P0, nullptr,
      Bn,  8,  8,  4,  4, 256, 3, 3, 2, 0, 1, 1);
  // 1x1 projection to latent z (no relu, f32 output for the quantizer)
  packw(ew5, 1, 256, 512);
  k_conv_wmma<256, 2><<<dim3(Bn*4*4/16, 16), 32, 0, stream>>>(P0, WP, eb5, nullptr, ZB,
      Bn,  4,  4,  4,  4, 512, 1, 1, 1, 0, 1, 0);

  // --- vector quantizer ---
  k_row_sumsq<<<4096, 32, 0, stream>>>(ZB, FN2, 512);
  k_col_sumsq<<<(1024 + 255) / 256, 256, 0, stream>>>(cb, EN2, 512, 1024);
  k_pack_w<<<(512 * 1024 + 255) / 256, 256, 0, stream>>>(cb, CBT, 1, 512, 1024);
  {
    int n = 4096 * 512;
    k_f32_to_f16<<<(n + 255) / 256, 256, 0, stream>>>(ZB, Z16, n);
  }
  k_vq_wmma<<<256, 32, 0, stream>>>(Z16, CBT, FN2, EN2, P0, idxO);

  // --- decoder (conv_transpose s=2 'SAME': lhs_dilation=2, pad_a=2) ---
  packw(dw1, 9, 512, 256);
  k_conv_wmma<512, 2><<<dim3(Bn*8*8/16, 8), 32, 0, stream>>>(P0, WP, db1, P1, nullptr,
      Bn,  4,  4,  8,  8, 256, 3, 3, 1, 2, 2, 1);
  packw(dw2, 9, 256, 128);
  k_conv_wmma<256, 2><<<dim3(Bn*16*16/16, 4), 32, 0, stream>>>(P1, WP, db2, P0, nullptr,
      Bn,  8,  8, 16, 16, 128, 3, 3, 1, 2, 2, 1);
  packw(dw3, 9, 128, 64);
  k_conv_wmma<128, 2><<<dim3(Bn*32*32/16, 2), 32, 0, stream>>>(P0, WP, db3, P1, nullptr,
      Bn, 16, 16, 32, 32,  64, 3, 3, 1, 2, 2, 1);
  packw(dw4, 9, 64, 32);
  k_conv_wmma<64, 2><<<dim3(Bn*64*64/16, 1), 32, 0, stream>>>(P1, WP, db4, P0, nullptr,
      Bn, 32, 32, 64, 64,  32, 3, 3, 1, 2, 2, 1);
  // final conv_transpose s=1 'SAME' == plain pad-1 conv, no relu, f32 out
  packw(dw5, 9, 32, 3);
  k_conv_wmma<32, 1><<<dim3(Bn*64*64/16, 1), 32, 0, stream>>>(P0, WP, db5, nullptr, recon,
      Bn, 64, 64, 64, 64,   3, 3, 3, 1, 1, 1, 0);
}
